// SetAbstraction_60833916780804
// MI455X (gfx1250) — matrix operand order repaired and checked by reading, not compiled, not run on hardware
//
#include <hip/hip_runtime.h>
#include <hip/hip_bf16.h>

typedef __attribute__((ext_vector_type(16))) __bf16       v16bf;
typedef __attribute__((ext_vector_type(8)))  float        v8f;
typedef __attribute__((ext_vector_type(4)))  unsigned int v4u;
typedef __attribute__((ext_vector_type(8)))  int          v8i;
typedef __attribute__((ext_vector_type(4)))  int          v4i;

#define B_  16
#define N_  32768
#define C_  64
#define S_  1024
#define K_  32
#define O_  1024
#define KC_ 2048

__device__ __forceinline__ unsigned short f2bf(float f) {
    unsigned int u = __float_as_uint(f);
    unsigned int r = u + 0x7fffu + ((u >> 16) & 1u);   // round-to-nearest-even
    return (unsigned short)(r >> 16);
}
__device__ __forceinline__ float bf2f(unsigned short h) {
    return __uint_as_float(((unsigned int)h) << 16);
}

union FragBF { v16bf v; unsigned int u[8]; };

// ---------------------------------------------------------------------------
// Tensor Data Mover: 2D bf16 tile  global -> LDS  (D# per ISA 08 §8.3/8.4)
//   group0: [1:0]=count=1, [63:32]=lds_addr, [120:64]=global_addr, [127:126]=2
//   group1: [17:16]=data_size(1 => 2B), [79:48]=tensor_dim0, [111:80]=tensor_dim1,
//           [127:112]=tile_dim0, [143:128]=tile_dim1, [207:160]=dim0_stride
//   builtin here is the 6-arg (clang-23) form:
//     (v4u g0, v8i g1, v4i g2, v4i g3, v8i extra, i32 cpol)
// ---------------------------------------------------------------------------
__device__ __forceinline__ void tdm_load_tile_bf16(
    unsigned short* lds_ptr, const unsigned short* gptr,
    unsigned int tensor_d0, unsigned int tensor_d1,
    unsigned int tile_d0, unsigned int tile_d1,
    unsigned int stride0, int lane)
{
#if __has_builtin(__builtin_amdgcn_tensor_load_to_lds)
    unsigned long long ga = (unsigned long long)gptr;
    unsigned int lds_addr = (unsigned int)(unsigned long long)(void*)lds_ptr;
    v4u g0;
    g0.x = 1u;                                           // count=1 (valid)
    g0.y = lds_addr;
    g0.z = (unsigned int)ga;
    g0.w = (unsigned int)((ga >> 32) & 0x1ffffffu) | (2u << 30);  // type=2
    v8i g1;
    g1[0] = (int)(1u << 16);                             // data_size = 2 bytes
    g1[1] = (int)((tensor_d0 & 0xffffu) << 16);          // tensor_dim0[15:0]
    g1[2] = (int)((tensor_d0 >> 16) | ((tensor_d1 & 0xffffu) << 16));
    g1[3] = (int)((tensor_d1 >> 16) | (tile_d0 << 16));  // tile_dim0
    g1[4] = (int)tile_d1;                                // tile_dim1 (tile_dim2=0)
    g1[5] = (int)stride0;                                // dim0 stride lo32
    g1[6] = 0;
    g1[7] = 0;
    v4i z4 = {0, 0, 0, 0};
    v8i z8 = {0, 0, 0, 0, 0, 0, 0, 0};
    __builtin_amdgcn_tensor_load_to_lds(g0, g1, z4, z4, z8, 0);
    (void)lane;
#else
    for (unsigned int i = (unsigned int)lane; i < tile_d0 * tile_d1; i += 32)
        lds_ptr[i] = gptr[(size_t)(i / tile_d0) * stride0 + (i % tile_d0)];
#endif
}

__device__ __forceinline__ void tdm_wait0() {
#if __has_builtin(__builtin_amdgcn_s_wait_tensorcnt)
    __builtin_amdgcn_s_wait_tensorcnt(0);
#endif
}

// ---------------------------------------------------------------------------
// 0) one-time fp32 -> bf16 hi/lo split of a weight matrix
// ---------------------------------------------------------------------------
__global__ void k_cvt_w(const float* __restrict__ W,
                        unsigned short* __restrict__ hi,
                        unsigned short* __restrict__ lo, int n) {
    int t = blockIdx.x * 256 + threadIdx.x;
    if (t >= n) return;
    float f = W[t];
    unsigned short h = f2bf(f);
    hi[t] = h;
    lo[t] = f2bf(f - bf2f(h));
}

// ---------------------------------------------------------------------------
// 1) sampled_x = x[:, sample_idx, :]   (second half of d_out)
// ---------------------------------------------------------------------------
__global__ void k_gather_sampled(const float* __restrict__ x,
                                 const int* __restrict__ sidx,
                                 float* __restrict__ out) {
    int t = blockIdx.x * 256 + threadIdx.x;      // over B_*S_*C_
    int c = t & (C_ - 1);
    int s = (t >> 6) & (S_ - 1);
    int b = t >> 16;
    out[t] = x[(size_t)b * N_ * C_ + (size_t)sidx[s] * C_ + c];
}

// ---------------------------------------------------------------------------
// 2) exact brute-force kNN (K=32), 1 WG/query, per-thread top-32 lists in LDS
// ---------------------------------------------------------------------------
__global__ void __launch_bounds__(128) k_knn(const float* __restrict__ x0,
                                             const int* __restrict__ sidx,
                                             int* __restrict__ nbr) {
    __shared__ float q[C_];
    __shared__ unsigned long long list[128 * 32];
    __shared__ unsigned long long tkey[128];
    const int tid = threadIdx.x;
    const int qi  = blockIdx.x;

    if (tid < C_) q[tid] = x0[(size_t)sidx[qi] * C_ + tid];
    unsigned long long* my = &list[tid * 32];
    #pragma unroll
    for (int i = 0; i < 32; ++i) my[i] = ~0ULL;
    __syncthreads();

    float qr[C_];
    #pragma unroll
    for (int i = 0; i < C_; ++i) qr[i] = q[i];

    unsigned long long curmax = ~0ULL; int curslot = 0;
    for (int n = tid; n < N_; n += 128) {
        const float4* p = (const float4*)(x0 + (size_t)n * C_);
        float d = 0.f;
        #pragma unroll
        for (int j = 0; j < 16; ++j) {
            float4 v = p[j];
            float d0 = v.x - qr[4 * j + 0];
            float d1 = v.y - qr[4 * j + 1];
            float d2 = v.z - qr[4 * j + 2];
            float d3 = v.w - qr[4 * j + 3];
            d += d0 * d0 + d1 * d1 + d2 * d2 + d3 * d3;
        }
        unsigned long long key =
            ((unsigned long long)__float_as_uint(d) << 32) | (unsigned int)n;
        if (key < curmax) {
            my[curslot] = key;
            unsigned long long m = 0; int ms = 0;
            #pragma unroll
            for (int i = 0; i < 32; ++i) {
                unsigned long long v = my[i];
                if (v > m) { m = v; ms = i; }
            }
            curmax = m; curslot = ms;
        }
    }
    __syncthreads();

    for (int r = 0; r < K_; ++r) {
        unsigned long long lm = ~0ULL; int ls = 0;
        #pragma unroll
        for (int i = 0; i < 32; ++i) {
            unsigned long long v = my[i];
            if (v < lm) { lm = v; ls = i; }
        }
        tkey[tid] = lm;
        __syncthreads();
        for (int off = 64; off > 0; off >>= 1) {
            if (tid < off) {
                unsigned long long o = tkey[tid + off];
                if (o < tkey[tid]) tkey[tid] = o;
            }
            __syncthreads();
        }
        unsigned long long g = tkey[0];
        if (lm == g) my[ls] = ~0ULL;
        if (tid == 0) nbr[qi * K_ + r] = (int)(g & 0xffffffffu);
        __syncthreads();
    }
}

// ---------------------------------------------------------------------------
// 3) GEMM1 (per batch): hT = relu(W1 @ gather(x,nbr) + b1), stored TRANSPOSED
//    as bf16 hi/lo: hT[kc][o] (2048 x 1024). A tiles via TDM (double-buffered),
//    B gathered+converted manually. 3x v_wmma per hi/lo pair.
// ---------------------------------------------------------------------------
__global__ void __launch_bounds__(256) k_gemm1(
    const float* __restrict__ x, const int* __restrict__ nbr,
    const unsigned short* __restrict__ W1hi, const unsigned short* __restrict__ W1lo,
    const float* __restrict__ b1,
    unsigned short* __restrict__ hThi, unsigned short* __restrict__ hTlo, int b)
{
    __shared__ __align__(16) unsigned short As[2][2][128][32]; // [buf][hi/lo][m][k]
    __shared__ __align__(16) unsigned short Bs[2][128][32];    // [hi/lo][n][k]
    const int tid  = threadIdx.x;
    const int lane = tid & 31;
    const int w    = tid >> 5;
    const int wm   = w & 3;
    const int wn   = w >> 2;
    const int n0   = blockIdx.x * 128;   // kc tile
    const int kg   = blockIdx.x * 2;
    const int m0   = blockIdx.y * 128;   // o tile

    v8f acc[2][4];
    v8f zero = {0.f, 0.f, 0.f, 0.f, 0.f, 0.f, 0.f, 0.f};
    #pragma unroll
    for (int i = 0; i < 2; ++i)
        #pragma unroll
        for (int j = 0; j < 4; ++j) acc[i][j] = zero;

    const int mrow = lane & 15;
    const int kbA  = (lane < 16) ? 0 : 16;
    const int kbB  = (lane < 16) ? 0 : 16;

    if (w == 0) {   // prologue: DMA first A tiles (k-step 0) into buffer 0
        tdm_load_tile_bf16(&As[0][0][0][0], W1hi + (size_t)m0 * S_,
                           S_, O_, 32, 128, S_, lane);
        tdm_load_tile_bf16(&As[0][1][0][0], W1lo + (size_t)m0 * S_,
                           S_, O_, 32, 128, S_, lane);
    }

    for (int kt = 0; kt < 32; ++kt) {
        const int s0  = kt * 32;
        const int cur = kt & 1;
        // --- B tile (fused gather): B[s, k*64+c] = x[b, nbr[s,k], c] ---
        {
            int rr = tid >> 2;
            int sl = rr & 31;
            int kl = rr >> 5;
            int c0 = (tid & 3) * 16;
            int pidx = nbr[(s0 + sl) * K_ + kg + kl];
            const float* src = x + (size_t)b * N_ * C_ + (size_t)pidx * C_ + c0;
            #pragma unroll
            for (int j = 0; j < 16; ++j) {
                float f = src[j];
                unsigned short hi = f2bf(f);
                unsigned short lo = f2bf(f - bf2f(hi));
                int n = kl * 64 + c0 + j;
                Bs[0][n][sl] = hi; Bs[1][n][sl] = lo;
            }
        }
        if (w == 0) tdm_wait0();          // A tile kt landed in LDS
        __syncthreads();
        if (w == 0 && kt + 1 < 32) {      // DMA next A tiles while we compute
            const size_t g = (size_t)m0 * S_ + (size_t)(kt + 1) * 32;
            tdm_load_tile_bf16(&As[cur ^ 1][0][0][0], W1hi + g, S_, O_, 32, 128, S_, lane);
            tdm_load_tile_bf16(&As[cur ^ 1][1][0][0], W1lo + g, S_, O_, 32, 128, S_, lane);
        }

        FragBF ah[2], al[2];
        #pragma unroll
        for (int tm = 0; tm < 2; ++tm) {
            int row = wm * 32 + tm * 16 + mrow;
            #pragma unroll
            for (int j = 0; j < 4; ++j) {
                ah[tm].u[j]     = *(const unsigned int*)&As[cur][0][row][kbA + 2 * j];
                ah[tm].u[j + 4] = *(const unsigned int*)&As[cur][0][row][8 + kbA + 2 * j];
                al[tm].u[j]     = *(const unsigned int*)&As[cur][1][row][kbA + 2 * j];
                al[tm].u[j + 4] = *(const unsigned int*)&As[cur][1][row][8 + kbA + 2 * j];
            }
        }
        #pragma unroll
        for (int tn = 0; tn < 4; ++tn) {
            FragBF bh, bl;
            int col = wn * 64 + tn * 16 + mrow;
            #pragma unroll
            for (int j = 0; j < 8; ++j) {
                bh.u[j] = *(const unsigned int*)&Bs[0][col][kbB + 2 * j];
                bl.u[j] = *(const unsigned int*)&Bs[1][col][kbB + 2 * j];
            }
            #pragma unroll
            for (int tm = 0; tm < 2; ++tm) {
                acc[tm][tn] = __builtin_amdgcn_wmma_f32_16x16x32_bf16(
                    false, ah[tm].v, false, bh.v, (short)0, acc[tm][tn], false, false);
                acc[tm][tn] = __builtin_amdgcn_wmma_f32_16x16x32_bf16(
                    false, ah[tm].v, false, bl.v, (short)0, acc[tm][tn], false, false);
                acc[tm][tn] = __builtin_amdgcn_wmma_f32_16x16x32_bf16(
                    false, al[tm].v, false, bh.v, (short)0, acc[tm][tn], false, false);
            }
        }
        __syncthreads();
    }

    // epilogue: +b1, relu, bf16 hi/lo split, LDS transpose, coalesced hT store
    unsigned short (*T)[128] = (unsigned short (*)[128])&As[0][0][0][0];
    const int rowl = tid >> 1;
    const int half = (tid & 1) * 64;
    #pragma unroll
    for (int hl = 0; hl < 2; ++hl) {
        #pragma unroll
        for (int tm = 0; tm < 2; ++tm)
            #pragma unroll
            for (int tn = 0; tn < 4; ++tn)
                #pragma unroll
                for (int r = 0; r < 8; ++r) {
                    int M  = (lane < 16) ? r : (r + 8);
                    int ol = wm * 32 + tm * 16 + M;
                    float v = acc[tm][tn][r] + b1[m0 + ol];
                    v = v > 0.f ? v : 0.f;
                    unsigned short h = f2bf(v);
                    unsigned short e = hl ? f2bf(v - bf2f(h)) : h;
                    T[wn * 64 + tn * 16 + (lane & 15)][ol] = e;
                }
        __syncthreads();
        unsigned short* dst = (hl ? hTlo : hThi) +
                              (size_t)(n0 + rowl) * O_ + m0 + half;
        const uint4* srcv = (const uint4*)&T[rowl][half];
        #pragma unroll
        for (int j = 0; j < 8; ++j) ((uint4*)dst)[j] = srcv[j];
        __syncthreads();
    }
}

// ---------------------------------------------------------------------------
// 4) GEMM2 (per batch, per k-slice): f[k] = W2 @ h[:, k*64..+64].
//    Both A (W2 bf16) and B (hT bf16, fragment-ready [n][k]) via TDM,
//    double-buffered.
// ---------------------------------------------------------------------------
__global__ void __launch_bounds__(256) k_gemm2(
    const unsigned short* __restrict__ hThi, const unsigned short* __restrict__ hTlo,
    const unsigned short* __restrict__ W2hi, const unsigned short* __restrict__ W2lo,
    float* __restrict__ f)
{
    __shared__ __align__(16) unsigned short As[2][2][128][32]; // [buf][hi/lo][m][k]
    __shared__ __align__(16) unsigned short Bs[2][2][64][32];  // [buf][hi/lo][n][k]
    const int tid  = threadIdx.x;
    const int lane = tid & 31;
    const int w    = tid >> 5;
    const int wm   = w & 3;
    const int wn   = w >> 2;
    const int kk   = blockIdx.x;     // k value 0..31
    const int m0   = blockIdx.y * 128;

    v8f acc[2][2];
    v8f zero = {0.f, 0.f, 0.f, 0.f, 0.f, 0.f, 0.f, 0.f};
    #pragma unroll
    for (int i = 0; i < 2; ++i)
        #pragma unroll
        for (int j = 0; j < 2; ++j) acc[i][j] = zero;

    const int mrow = lane & 15;
    const int kbA  = (lane < 16) ? 0 : 16;
    const int kbB  = (lane < 16) ? 0 : 16;
    const size_t hbase = (size_t)(kk * 64) * O_;

    if (w == 0) {
        tdm_load_tile_bf16(&As[0][0][0][0], W2hi + (size_t)m0 * O_, O_, O_, 32, 128, O_, lane);
        tdm_load_tile_bf16(&As[0][1][0][0], W2lo + (size_t)m0 * O_, O_, O_, 32, 128, O_, lane);
        tdm_load_tile_bf16(&Bs[0][0][0][0], hThi + hbase, O_, KC_, 32, 64, O_, lane);
        tdm_load_tile_bf16(&Bs[0][1][0][0], hTlo + hbase, O_, KC_, 32, 64, O_, lane);
    }

    for (int kt = 0; kt < 32; ++kt) {
        const int cur = kt & 1;
        if (w == 0) tdm_wait0();
        __syncthreads();
        if (w == 0 && kt + 1 < 32) {
            const size_t s1 = (size_t)(kt + 1) * 32;
            tdm_load_tile_bf16(&As[cur ^ 1][0][0][0], W2hi + (size_t)m0 * O_ + s1,
                               O_, O_, 32, 128, O_, lane);
            tdm_load_tile_bf16(&As[cur ^ 1][1][0][0], W2lo + (size_t)m0 * O_ + s1,
                               O_, O_, 32, 128, O_, lane);
            tdm_load_tile_bf16(&Bs[cur ^ 1][0][0][0], hThi + hbase + s1,
                               O_, KC_, 32, 64, O_, lane);
            tdm_load_tile_bf16(&Bs[cur ^ 1][1][0][0], hTlo + hbase + s1,
                               O_, KC_, 32, 64, O_, lane);
        }

        FragBF ah[2], al[2];
        #pragma unroll
        for (int tm = 0; tm < 2; ++tm) {
            int row = wm * 32 + tm * 16 + mrow;
            #pragma unroll
            for (int j = 0; j < 4; ++j) {
                ah[tm].u[j]     = *(const unsigned int*)&As[cur][0][row][kbA + 2 * j];
                ah[tm].u[j + 4] = *(const unsigned int*)&As[cur][0][row][8 + kbA + 2 * j];
                al[tm].u[j]     = *(const unsigned int*)&As[cur][1][row][kbA + 2 * j];
                al[tm].u[j + 4] = *(const unsigned int*)&As[cur][1][row][8 + kbA + 2 * j];
            }
        }
        #pragma unroll
        for (int tn = 0; tn < 2; ++tn) {
            FragBF bh, bl;
            int col = wn * 32 + tn * 16 + mrow;
            #pragma unroll
            for (int j = 0; j < 8; ++j) {
                bh.u[j] = *(const unsigned int*)&Bs[cur][0][col][kbB + 2 * j];
                bl.u[j] = *(const unsigned int*)&Bs[cur][1][col][kbB + 2 * j];
            }
            #pragma unroll
            for (int tm = 0; tm < 2; ++tm) {
                acc[tm][tn] = __builtin_amdgcn_wmma_f32_16x16x32_bf16(
                    false, ah[tm].v, false, bh.v, (short)0, acc[tm][tn], false, false);
                acc[tm][tn] = __builtin_amdgcn_wmma_f32_16x16x32_bf16(
                    false, ah[tm].v, false, bl.v, (short)0, acc[tm][tn], false, false);
                acc[tm][tn] = __builtin_amdgcn_wmma_f32_16x16x32_bf16(
                    false, al[tm].v, false, bh.v, (short)0, acc[tm][tn], false, false);
            }
        }
        __syncthreads();
    }

    #pragma unroll
    for (int tm = 0; tm < 2; ++tm)
        #pragma unroll
        for (int tn = 0; tn < 2; ++tn)
            #pragma unroll
            for (int r = 0; r < 8; ++r) {
                int M = (lane < 16) ? r : (r + 8);
                int p = m0 + wm * 32 + tm * 16 + M;
                int c = wn * 32 + tn * 16 + (lane & 15);
                f[(size_t)kk * S_ * C_ + (size_t)p * C_ + c] = acc[tm][tn][r];
            }
}

// ---------------------------------------------------------------------------
// 5) max over k, then +b2 (bias is k-invariant)
// ---------------------------------------------------------------------------
__global__ void k_maxk(const float* __restrict__ f,
                       const float* __restrict__ b2,
                       float* __restrict__ out) {
    int t = blockIdx.x * 256 + threadIdx.x;
    int p = t >> 6;
    float m = -3.402823466e38f;
    #pragma unroll
    for (int k = 0; k < 32; ++k) {
        float v = f[(size_t)k * S_ * C_ + t];
        m = v > m ? v : m;
    }
    out[t] = m + b2[p];
}

// ---------------------------------------------------------------------------
extern "C" void kernel_launch(void* const* d_in, const int* in_sizes, int n_in,
                              void* d_out, int out_size, void* d_ws, size_t ws_size,
                              hipStream_t stream) {
    (void)in_sizes; (void)n_in; (void)out_size; (void)ws_size;
    const float* x    = (const float*)d_in[0];
    const int*   sidx = (const int*)d_in[1];
    const float* W1   = (const float*)d_in[2];
    const float* b1   = (const float*)d_in[3];
    const float* W2   = (const float*)d_in[4];
    const float* b2   = (const float*)d_in[5];

    float* feats   = (float*)d_out;
    float* sampled = feats + (size_t)B_ * S_ * C_;

    char* ws = (char*)d_ws;
    int*            nbr  = (int*)ws;                               // 128 KB
    unsigned short* W1hi = (unsigned short*)(ws + 131072);         // 2 MB each
    unsigned short* W1lo = W1hi + (size_t)O_ * S_;
    unsigned short* W2hi = W1lo + (size_t)O_ * S_;
    unsigned short* W2lo = W2hi + (size_t)O_ * O_;
    unsigned short* hThi = W2lo + (size_t)O_ * O_;                 // 4 MB each
    unsigned short* hTlo = hThi + (size_t)KC_ * O_;
    float*          f    = (float*)(hTlo + (size_t)KC_ * O_);      // 8 MB

    k_cvt_w<<<(O_ * S_) / 256, 256, 0, stream>>>(W1, W1hi, W1lo, O_ * S_);
    k_cvt_w<<<(O_ * O_) / 256, 256, 0, stream>>>(W2, W2hi, W2lo, O_ * O_);
    k_gather_sampled<<<(B_ * S_ * C_) / 256, 256, 0, stream>>>(x, sidx, sampled);
    k_knn<<<S_, 128, 0, stream>>>(x, sidx, nbr);

    for (int b = 0; b < B_; ++b) {
        k_gemm1<<<dim3(16, 8), 256, 0, stream>>>(x, nbr, W1hi, W1lo, b1, hThi, hTlo, b);
        k_gemm2<<<dim3(32, 8), 256, 0, stream>>>(hThi, hTlo, W2hi, W2lo, f);
        k_maxk<<<(S_ * C_) / 256, 256, 0, stream>>>(f, b2, feats + (size_t)b * S_ * C_);
    }
}
